// HI_Attention_86139864088752
// MI455X (gfx1250) — matrix operand order repaired and checked by reading, not compiled, not run on hardware
//
#include <hip/hip_runtime.h>
#include <cmath>

// ---------------- problem constants ----------------
#define BB   16
#define NN   785
#define CC   768
#define HH   12
#define HD   64
#define TT   797          // N + H
#define TP   800          // padded T (multiple of 16)
#define K3   2304         // 3*C
#define LN_EPS 1e-5f

typedef __attribute__((ext_vector_type(16))) __bf16 v16bf;
typedef __attribute__((ext_vector_type(8)))  float  v8f;

union ABFrag {
    uint4 q[2];
    v16bf v;
};

// ---------------------------------------------------------------------------
// 1) convert x (f32, B,N,C) into xa rows [0..N) per batch (bf16, B,T,C)
// ---------------------------------------------------------------------------
__global__ void k_convert_x(const float* __restrict__ x, __bf16* __restrict__ xa) {
    size_t idx = (size_t)blockIdx.x * blockDim.x + threadIdx.x;
    size_t total = (size_t)BB * NN * CC;
    if (idx >= total) return;
    int c = (int)(idx % CC);
    size_t bn = idx / CC;
    int n = (int)(bn % NN);
    int b = (int)(bn / NN);
    xa[((size_t)b * TT + n) * CC + c] = (__bf16)x[idx];
}

// ---------------------------------------------------------------------------
// 2) head tokens: mean-pool -> ht_proj -> LayerNorm(hd) -> gelu -> +pos
//    one block per (b,h); writes xa row b*T + N + h (bf16)
// ---------------------------------------------------------------------------
__global__ __launch_bounds__(256)
void k_head_tokens(const float* __restrict__ x, const float* __restrict__ htp_w,
                   const float* __restrict__ htp_b, const float* __restrict__ ln_g,
                   const float* __restrict__ ln_b, const float* __restrict__ pos,
                   __bf16* __restrict__ xa) {
    int bh = blockIdx.x;
    int h = bh % HH, b = bh / HH;
    int tid = threadIdx.x;

    __shared__ float part[256];
    __shared__ float xh[HD];
    __shared__ float y[CC];
    __shared__ float cmean[HH], crstd[HH];

    int d = tid & 63, grp = tid >> 6;          // 4 groups of 64
    float s = 0.0f;
    for (int n = grp; n < NN; n += 4)
        s += x[((size_t)b * NN + n) * CC + h * HD + d];
    part[tid] = s;
    __syncthreads();
    if (tid < HD)
        xh[tid] = (part[tid] + part[tid + 64] + part[tid + 128] + part[tid + 192]) * (1.0f / NN);
    __syncthreads();

    for (int o = tid; o < CC; o += 256) {
        float acc = htp_b[o];
        #pragma unroll 8
        for (int kk = 0; kk < HD; ++kk) acc += xh[kk] * htp_w[kk * CC + o];
        y[o] = acc;
    }
    __syncthreads();

    if (tid < HH) {                            // per-64 chunk LN stats
        float m = 0.0f;
        for (int j = 0; j < HD; ++j) m += y[tid * HD + j];
        m *= (1.0f / HD);
        float v = 0.0f;
        for (int j = 0; j < HD; ++j) { float dd = y[tid * HD + j] - m; v += dd * dd; }
        v *= (1.0f / HD);
        cmean[tid] = m;
        crstd[tid] = rsqrtf(v + LN_EPS);
    }
    __syncthreads();

    for (int o = tid; o < CC; o += 256) {
        int ch = o >> 6, j = o & 63;
        float t = (y[o] - cmean[ch]) * crstd[ch] * ln_g[j] + ln_b[j];
        float g = 0.5f * t * (1.0f + erff(t * 0.70710678118654752f));   // exact gelu
        g += pos[h * CC + o];
        xa[((size_t)b * TT + NN + h) * CC + o] = (__bf16)g;
    }
}

// ---------------------------------------------------------------------------
// 3) weight transpose + f32->bf16:  in (K,Nn) row-major  ->  out (Nn,K)
// ---------------------------------------------------------------------------
__global__ void k_transpose_w(const float* __restrict__ in, __bf16* __restrict__ out,
                              int K, int Nn) {
    size_t idx = (size_t)blockIdx.x * blockDim.x + threadIdx.x;
    size_t total = (size_t)K * Nn;
    if (idx >= total) return;
    int o = (int)(idx % Nn);
    int c = (int)(idx / Nn);
    out[(size_t)o * K + c] = (__bf16)in[idx];
}

// ---------------------------------------------------------------------------
// 4) bf16 GEMM with 2x4 register blocking: each wave computes a 32x64 macro
//    tile (8 WMMAs per k-step from 6 fragment loads). C = A(MxK)*Bt(NxK)^T.
// ---------------------------------------------------------------------------
template <bool OUT_BF16>
__global__ __launch_bounds__(128)
void k_gemm(const __bf16* __restrict__ A, const __bf16* __restrict__ Bt,
            const float* __restrict__ bias, void* __restrict__ Cout,
            int M, int Nn, int K, int Mb, int Nb) {
    int wave = blockIdx.x * 4 + (threadIdx.x >> 5);
    if (wave >= Mb * Nb) return;
    int lane = threadIdx.x & 31;
    int mb = wave / Nb, nb = wave % Nb;
    int r = lane & 15, half = lane >> 4;

    int m0 = mb * 32 + r;
    int m1 = m0 + 16;
    int m0c = m0 < M ? m0 : M - 1;             // clamp edge loads (stores guarded)
    int m1c = m1 < M ? m1 : M - 1;

    const __bf16* arow0 = A + (size_t)m0c * K + half * 8;
    const __bf16* arow1 = A + (size_t)m1c * K + half * 8;
    const __bf16* brow0 = Bt + (size_t)(nb * 64 + r) * K + half * 8;
    const __bf16* brow1 = brow0 + (size_t)16 * K;
    const __bf16* brow2 = brow0 + (size_t)32 * K;
    const __bf16* brow3 = brow0 + (size_t)48 * K;

    v8f acc[2][4] = {};
    for (int k = 0; k < K; k += 32) {
        ABFrag a0, a1, b0, b1, b2, b3;
        a0.q[0] = *(const uint4*)(arow0 + k);  a0.q[1] = *(const uint4*)(arow0 + k + 16);
        a1.q[0] = *(const uint4*)(arow1 + k);  a1.q[1] = *(const uint4*)(arow1 + k + 16);
        b0.q[0] = *(const uint4*)(brow0 + k);  b0.q[1] = *(const uint4*)(brow0 + k + 16);
        b1.q[0] = *(const uint4*)(brow1 + k);  b1.q[1] = *(const uint4*)(brow1 + k + 16);
        b2.q[0] = *(const uint4*)(brow2 + k);  b2.q[1] = *(const uint4*)(brow2 + k + 16);
        b3.q[0] = *(const uint4*)(brow3 + k);  b3.q[1] = *(const uint4*)(brow3 + k + 16);
        __builtin_prefetch(arow0 + k + 128, 0, 1);   // global_prefetch_b8
        __builtin_prefetch(brow0 + k + 128, 0, 1);

        acc[0][0] = __builtin_amdgcn_wmma_f32_16x16x32_bf16(false, a0.v, false, b0.v,
                                                            (short)0, acc[0][0], false, false);
        acc[0][1] = __builtin_amdgcn_wmma_f32_16x16x32_bf16(false, a0.v, false, b1.v,
                                                            (short)0, acc[0][1], false, false);
        acc[0][2] = __builtin_amdgcn_wmma_f32_16x16x32_bf16(false, a0.v, false, b2.v,
                                                            (short)0, acc[0][2], false, false);
        acc[0][3] = __builtin_amdgcn_wmma_f32_16x16x32_bf16(false, a0.v, false, b3.v,
                                                            (short)0, acc[0][3], false, false);
        acc[1][0] = __builtin_amdgcn_wmma_f32_16x16x32_bf16(false, a1.v, false, b0.v,
                                                            (short)0, acc[1][0], false, false);
        acc[1][1] = __builtin_amdgcn_wmma_f32_16x16x32_bf16(false, a1.v, false, b1.v,
                                                            (short)0, acc[1][1], false, false);
        acc[1][2] = __builtin_amdgcn_wmma_f32_16x16x32_bf16(false, a1.v, false, b2.v,
                                                            (short)0, acc[1][2], false, false);
        acc[1][3] = __builtin_amdgcn_wmma_f32_16x16x32_bf16(false, a1.v, false, b3.v,
                                                            (short)0, acc[1][3], false, false);
    }

    #pragma unroll
    for (int j = 0; j < 4; ++j) {
        int col = nb * 64 + j * 16 + r;
        float bval = bias ? bias[col] : 0.0f;
        #pragma unroll
        for (int i = 0; i < 2; ++i) {
            int row0 = mb * 32 + i * 16 + half * 8;
            #pragma unroll
            for (int v = 0; v < 8; ++v) {
                int row = row0 + v;
                if (row < M) {
                    float val = acc[i][j][v] + bval;
                    if (OUT_BF16) ((__bf16*)Cout)[(size_t)row * Nn + col] = (__bf16)val;
                    else          ((float*)Cout)[(size_t)row * Nn + col]  = val;
                }
            }
        }
    }
}

// ---------------------------------------------------------------------------
// 5) split qkv (B*T x 2304 bf16) -> q,k (b,h,Tp,64) and v transposed (b,h,64,Tp)
//    zero-fill the Tp padding so WMMA never sees NaN garbage
// ---------------------------------------------------------------------------
__global__ void k_split_qkv(const __bf16* __restrict__ qkv, __bf16* __restrict__ q,
                            __bf16* __restrict__ k, __bf16* __restrict__ vt) {
    size_t idx = (size_t)blockIdx.x * blockDim.x + threadIdx.x;
    size_t total = (size_t)BB * HH * TP * HD;
    if (idx >= total) return;
    int d = (int)(idx & 63);
    size_t t1 = idx >> 6;
    int t = (int)(t1 % TP);
    size_t bh = t1 / TP;
    int h = (int)(bh % HH), b = (int)(bh / HH);

    __bf16 qv, kv, vv;
    if (t < TT) {
        const __bf16* src = qkv + ((size_t)b * TT + t) * K3;
        qv = src[0 * CC + h * HD + d];
        kv = src[1 * CC + h * HD + d];
        vv = src[2 * CC + h * HD + d];
    } else {
        qv = (__bf16)0.0f; kv = (__bf16)0.0f; vv = (__bf16)0.0f;
    }
    q [(bh * TP + t) * HD + d] = qv;
    k [(bh * TP + t) * HD + d] = kv;
    vt[(bh * HD + d) * TP + t] = vv;
}

// ---------------------------------------------------------------------------
// 6) attention: one block (4 waves) per (b,h,query-tile of 16)
//    phase1 S=QK^T*scale in LDS, phase2 masked softmax, phase3 O=P*V
// ---------------------------------------------------------------------------
__global__ __launch_bounds__(128)
void k_attn(const __bf16* __restrict__ q, const __bf16* __restrict__ k,
            const __bf16* __restrict__ vt, __bf16* __restrict__ outp) {
    constexpr int QT = TP / 16;                  // 50 key tiles
    int bid = blockIdx.x;
    int qt = bid % QT;
    int bh = bid / QT;
    int h = bh % HH, b = bh / HH;
    int tid = threadIdx.x, wave = tid >> 5, lane = tid & 31;
    int r = lane & 15, half = lane >> 4;

    __shared__ float S[16][TP];
    __shared__ float red[16][8];
    __shared__ float rowmax[16], rowrinv[16];

    const size_t bhrow = (size_t)bh * TP;

    // ---- phase 1: scores ----
    const __bf16* qptr = q + (bhrow + qt * 16 + r) * HD + half * 8;
    ABFrag a0, a1;
    a0.q[0] = *(const uint4*)(qptr + 0);   a0.q[1] = *(const uint4*)(qptr + 16);
    a1.q[0] = *(const uint4*)(qptr + 32);  a1.q[1] = *(const uint4*)(qptr + 48);

    for (int nt = wave; nt < QT; nt += 4) {
        const __bf16* kptr = k + (bhrow + nt * 16 + r) * HD + half * 8;
        ABFrag b0, b1;
        b0.q[0] = *(const uint4*)(kptr + 0);   b0.q[1] = *(const uint4*)(kptr + 16);
        b1.q[0] = *(const uint4*)(kptr + 32);  b1.q[1] = *(const uint4*)(kptr + 48);
        v8f acc = {};
        acc = __builtin_amdgcn_wmma_f32_16x16x32_bf16(false, a0.v, false, b0.v,
                                                      (short)0, acc, false, false);
        acc = __builtin_amdgcn_wmma_f32_16x16x32_bf16(false, a1.v, false, b1.v,
                                                      (short)0, acc, false, false);
        int kcol = nt * 16 + r;
        bool valid = kcol < TT;
        #pragma unroll
        for (int v = 0; v < 8; ++v)
            S[half * 8 + v][kcol] = valid ? acc[v] * 0.125f : -3.0e38f;
    }
    __syncthreads();

    // ---- phase 2: softmax (row = tid>>3, 8 threads per row) ----
    {
        int rr = tid >> 3, g = tid & 7;
        float m = -3.0e38f;
        for (int c = g; c < TP; c += 8) m = fmaxf(m, S[rr][c]);
        red[rr][g] = m;
        __syncthreads();
        if (g == 0) {
            float mm = red[rr][0];
            #pragma unroll
            for (int i = 1; i < 8; ++i) mm = fmaxf(mm, red[rr][i]);
            rowmax[rr] = mm;
        }
        __syncthreads();
        float mm = rowmax[rr];
        float ssum = 0.0f;
        for (int c = g; c < TP; c += 8) {
            float e = __expf(S[rr][c] - mm);
            S[rr][c] = e;
            ssum += e;
        }
        red[rr][g] = ssum;
        __syncthreads();
        if (g == 0) {
            float t = 0.0f;
            #pragma unroll
            for (int i = 0; i < 8; ++i) t += red[rr][i];
            rowrinv[rr] = 1.0f / t;
        }
        __syncthreads();
    }

    // ---- phase 3: O = P * V ; wave w owns channel tile w (hd=64 = 4 tiles) ----
    {
        int dt = wave;
        float rinv = rowrinv[r];
        const __bf16* vptr = vt + ((size_t)bh * HD + dt * 16 + r) * TP + half * 8;
        v8f acc = {};
        for (int kk = 0; kk < TP; kk += 32) {
            ABFrag av, bv;
            int base = kk + half * 8;
            #pragma unroll
            for (int e = 0; e < 8; ++e) av.v[e]     = (__bf16)(S[r][base + e] * rinv);
            #pragma unroll
            for (int e = 0; e < 8; ++e) av.v[8 + e] = (__bf16)(S[r][base + 16 + e] * rinv);
            bv.q[0] = *(const uint4*)(vptr + kk);
            bv.q[1] = *(const uint4*)(vptr + kk + 16);
            acc = __builtin_amdgcn_wmma_f32_16x16x32_bf16(false, av.v, false, bv.v,
                                                          (short)0, acc, false, false);
        }
        int col = h * HD + dt * 16 + r;
        #pragma unroll
        for (int v = 0; v < 8; ++v) {
            int t = qt * 16 + half * 8 + v;
            if (t < TT)
                outp[((size_t)b * TT + t) * CC + col] = (__bf16)acc[v];
        }
    }
}

// ---------------------------------------------------------------------------
// 7) final: cls = row0 + mean(head-token rows); rows 1..N-1 pass through
// ---------------------------------------------------------------------------
__global__ void k_final(const float* __restrict__ pout, float* __restrict__ out) {
    size_t idx = (size_t)blockIdx.x * blockDim.x + threadIdx.x;
    size_t total = (size_t)BB * NN * CC;
    if (idx >= total) return;
    int c = (int)(idx % CC);
    size_t bn = idx / CC;
    int n = (int)(bn % NN);
    int b = (int)(bn / NN);
    float v = pout[((size_t)b * TT + n) * CC + c];
    if (n == 0) {
        float s = 0.0f;
        #pragma unroll
        for (int hh = 0; hh < HH; ++hh)
            s += pout[((size_t)b * TT + NN + hh) * CC + c];
        v += s * (1.0f / HH);
    }
    out[idx] = v;
}

// ---------------------------------------------------------------------------
extern "C" void kernel_launch(void* const* d_in, const int* in_sizes, int n_in,
                              void* d_out, int out_size, void* d_ws, size_t ws_size,
                              hipStream_t stream) {
    (void)in_sizes; (void)n_in; (void)out_size; (void)ws_size;
    const float* x      = (const float*)d_in[0];
    const float* qkv_w  = (const float*)d_in[1];
    const float* proj_w = (const float*)d_in[2];
    const float* proj_b = (const float*)d_in[3];
    const float* htp_w  = (const float*)d_in[4];
    const float* htp_b  = (const float*)d_in[5];
    const float* ln_g   = (const float*)d_in[6];
    const float* ln_b   = (const float*)d_in[7];
    const float* pos    = (const float*)d_in[8];
    float* out = (float*)d_out;

    auto align256 = [](size_t v) { return (v + 255) & ~(size_t)255; };
    char* ws = (char*)d_ws;
    size_t off = 0;
    const size_t M = (size_t)BB * TT;                       // 12752 rows

    __bf16* xa      = (__bf16*)(ws + off); off = align256(off + M * CC * 2);
    __bf16* wqkv_t  = (__bf16*)(ws + off); off = align256(off + (size_t)K3 * CC * 2);
    __bf16* wproj_t = (__bf16*)(ws + off); off = align256(off + (size_t)CC * CC * 2);
    __bf16* qkv_bf  = (__bf16*)(ws + off); off = align256(off + M * K3 * 2);
    __bf16* q_bf    = (__bf16*)(ws + off); off = align256(off + (size_t)BB * HH * TP * HD * 2);
    __bf16* k_bf    = (__bf16*)(ws + off); off = align256(off + (size_t)BB * HH * TP * HD * 2);
    __bf16* vt_bf   = (__bf16*)(ws + off); off = align256(off + (size_t)BB * HH * TP * HD * 2);
    __bf16* aout    = (__bf16*)(ws + off); off = align256(off + M * CC * 2);
    float*  pout    = (float*)(ws + off);  off = align256(off + M * CC * 4);

    // 1) x -> bf16 xa rows
    {
        size_t total = (size_t)BB * NN * CC;
        k_convert_x<<<(unsigned)((total + 255) / 256), 256, 0, stream>>>(x, xa);
    }
    // 2) head tokens into xa tail rows
    k_head_tokens<<<BB * HH, 256, 0, stream>>>(x, htp_w, htp_b, ln_g, ln_b, pos, xa);
    // 3) weight transposes
    {
        size_t t1 = (size_t)CC * K3;
        k_transpose_w<<<(unsigned)((t1 + 255) / 256), 256, 0, stream>>>(qkv_w, wqkv_t, CC, K3);
        size_t t2 = (size_t)CC * CC;
        k_transpose_w<<<(unsigned)((t2 + 255) / 256), 256, 0, stream>>>(proj_w, wproj_t, CC, CC);
    }
    // 4) qkv GEMM (bf16 out, no bias): 32x64 macro tiles
    {
        int Mb = (int)((M + 31) / 32);          // 399
        int Nb = K3 / 64;                        // 36
        int waves = Mb * Nb;
        k_gemm<true><<<(waves + 3) / 4, 128, 0, stream>>>(xa, wqkv_t, nullptr, qkv_bf,
                                                          (int)M, K3, CC, Mb, Nb);
    }
    // 5) split into q / k / v^T with zero padding
    {
        size_t total = (size_t)BB * HH * TP * HD;
        k_split_qkv<<<(unsigned)((total + 255) / 256), 256, 0, stream>>>(qkv_bf, q_bf, k_bf, vt_bf);
    }
    // 6) attention
    k_attn<<<BB * HH * (TP / 16), 128, 0, stream>>>(q_bf, k_bf, vt_bf, aout);
    // 7) proj GEMM (f32 out + bias): 32x64 macro tiles
    {
        int Mb = (int)((M + 31) / 32);          // 399
        int Nb = CC / 64;                        // 12
        int waves = Mb * Nb;
        k_gemm<false><<<(waves + 3) / 4, 128, 0, stream>>>(aout, wproj_t, proj_b, pout,
                                                           (int)M, CC, CC, Mb, Nb);
    }
    // 8) cls merge -> output
    {
        size_t total = (size_t)BB * NN * CC;
        k_final<<<(unsigned)((total + 255) / 256), 256, 0, stream>>>(pout, out);
    }
}
